// FreeRC_56341380989596
// MI455X (gfx1250) — matrix operand order repaired and checked by reading, not compiled, not run on hardware
//
#include <hip/hip_runtime.h>
#include <hip/hip_bf16.h>
#include <math.h>

#define N_NODES 16384
#define D_FEAT  256

typedef __attribute__((ext_vector_type(16))) __bf16 v16bf;
typedef __attribute__((ext_vector_type(8)))  __bf16 v8bf;
typedef __attribute__((ext_vector_type(8)))  float  v8f;
typedef __attribute__((ext_vector_type(4)))  float  v4f;

// ---------------------------------------------------------------------------
// Prep: X^T as bf16  (XT[c][r] = X[r][c])  -> B-operand of phase-1 WMMA is
// K-contiguous.  Runs once; X stays L2-resident for the strided reads.
// ---------------------------------------------------------------------------
__global__ __launch_bounds__(256) void prep_xt(const float* __restrict__ X,
                                               __bf16* __restrict__ XT) {
  int i = blockIdx.x * blockDim.x + threadIdx.x;       // over 256*16384
  int c = i >> 14;                                     // feature col 0..255
  int r = i & (N_NODES - 1);                           // node row
  XT[i] = (__bf16)X[(size_t)r * D_FEAT + c];
}

// Prep: W_agg^T as bf16 (WT[n][k] = W_agg[k][n])
__global__ __launch_bounds__(256) void prep_wt(const float* __restrict__ W,
                                               __bf16* __restrict__ WT) {
  int i = blockIdx.x * blockDim.x + threadIdx.x;       // over 256*256
  int c = i >> 8;
  int r = i & (D_FEAT - 1);
  WT[i] = (__bf16)W[(size_t)r * D_FEAT + c];
}

// ---------------------------------------------------------------------------
// Fused:  out = sigmoid(w) * relu( (A_hat @ X) @ W_agg ) + (1-sigmoid(w)) * X
// Block = 512 threads = 16 waves; block handles 16 output rows x 256 cols.
// Phase 1: no barriers / no LDS; per-lane A fragments straight from global
// (one contiguous 64B f32 run per WMMA, NT-hinted), bf16 conversion in registers,
// DUAL accumulators to break the WMMA->WMMA RAW chain so conversion VALU of
// one chain fills the XDL hazard slots of the other.
// ---------------------------------------------------------------------------
#define T_PITCH 264   // bf16 elems per LDS T row (528 B: conflict-free b128)

__global__ __launch_bounds__(512)
void gcn_fused(const float*  __restrict__ A,    // [N][N] f32
               const float*  __restrict__ X,    // [N][D] f32
               const float*  __restrict__ wv,   // [N]    f32
               const __bf16* __restrict__ XT,   // [D][N] bf16
               const __bf16* __restrict__ WT,   // [D][D] bf16
               float*        __restrict__ out)  // [N][D] f32
{
  __shared__ alignas(16) __bf16 tTile[16 * T_PITCH];   // T = A_hat@X tile (bf16)

  const int tid  = threadIdx.x;
  const int wave = tid >> 5;          // 0..15
  const int lane = tid & 31;
  const int m    = lane & 15;         // row within 16 (A/C layout)
  const int half = lane >> 4;         // 0/1: K/M upper-half selector
  const int r0   = blockIdx.x * 16;   // output row base
  const int n0   = wave * 16;         // output col base for this wave
  const int n    = n0 + m;            // this lane's output column

  v8f c0 = {};                         // dual phase-1 accumulators
  v8f c1 = {};

  // per-lane streaming pointers
  const float*  ap = A  + (size_t)(r0 + m) * N_NODES + half * 16; // A row, run base
  const __bf16* bp = XT + (size_t)n * N_NODES + half * 16;        // X^T column n

  // ---------------- Phase 1: T = A_hat[r0:r0+16, :] @ X ----------------
#pragma unroll 4
  for (int k0 = 0; k0 < N_NODES; k0 += 64) {
    // ---- chunk 0: K in [k0, k0+32) -> accumulator c0 ----
    {
      v4f a0 = __builtin_nontemporal_load((const v4f*)(ap + k0));
      v4f a1 = __builtin_nontemporal_load((const v4f*)(ap + k0 + 4));
      v4f a2 = __builtin_nontemporal_load((const v4f*)(ap + k0 + 8));
      v4f a3 = __builtin_nontemporal_load((const v4f*)(ap + k0 + 12));
      v16bf b = *(const v16bf*)(bp + k0);
      v16bf a;
#pragma unroll
      for (int e = 0; e < 4; ++e) {
        a[e]      = (__bf16)a0[e];
        a[e + 4]  = (__bf16)a1[e];
        a[e + 8]  = (__bf16)a2[e];
        a[e + 12] = (__bf16)a3[e];
      }
      c0 = __builtin_amdgcn_wmma_f32_16x16x32_bf16(
          false, a, false, b, (short)0, c0, false, false);
    }
    // ---- chunk 1: K in [k0+32, k0+64) -> accumulator c1 ----
    {
      v4f a0 = __builtin_nontemporal_load((const v4f*)(ap + k0 + 32));
      v4f a1 = __builtin_nontemporal_load((const v4f*)(ap + k0 + 36));
      v4f a2 = __builtin_nontemporal_load((const v4f*)(ap + k0 + 40));
      v4f a3 = __builtin_nontemporal_load((const v4f*)(ap + k0 + 44));
      v16bf b = *(const v16bf*)(bp + k0 + 32);
      v16bf a;
#pragma unroll
      for (int e = 0; e < 4; ++e) {
        a[e]      = (__bf16)a0[e];
        a[e + 4]  = (__bf16)a1[e];
        a[e + 8]  = (__bf16)a2[e];
        a[e + 12] = (__bf16)a3[e];
      }
      c1 = __builtin_amdgcn_wmma_f32_16x16x32_bf16(
          false, a, false, b, (short)0, c1, false, false);
    }
  }

  // merge accumulators, scatter T tile (bf16) into LDS for cross-wave K access
#pragma unroll
  for (int j = 0; j < 8; ++j) {
    int row = j + 8 * half;            // C layout: VGPR j <-> row j + 8*half
    tTile[row * T_PITCH + n] = (__bf16)(c0[j] + c1[j]);
  }
  __syncthreads();

  // ---------------- Phase 2: out tile = relu(T @ W_agg) blended -------
  v8f c2 = {};
#pragma unroll
  for (int kk = 0; kk < D_FEAT; kk += 32) {
    v8bf ra0 = *(const v8bf*)&tTile[m * T_PITCH + kk + half * 16];
    v8bf ra1 = *(const v8bf*)&tTile[m * T_PITCH + kk + half * 16 + 8];
    v16bf a;
#pragma unroll
    for (int e = 0; e < 8; ++e) { a[e] = ra0[e]; a[e + 8] = ra1[e]; }

    v16bf b = *(const v16bf*)(WT + (size_t)n * D_FEAT + (kk + half * 16));

    c2 = __builtin_amdgcn_wmma_f32_16x16x32_bf16(
        false, a, false, b, (short)0, c2, false, false);
  }

  // epilogue: s*relu(agg) + (1-s)*X
#pragma unroll
  for (int j = 0; j < 8; ++j) {
    int r = r0 + j + 8 * half;
    float s   = 1.0f / (1.0f + __expf(-wv[r]));
    float agg = c2[j] > 0.0f ? c2[j] : 0.0f;
    float xv  = X[(size_t)r * D_FEAT + n];
    __builtin_nontemporal_store(s * agg + (1.0f - s) * xv,
                                &out[(size_t)r * D_FEAT + n]);
  }
}

// ---------------------------------------------------------------------------
extern "C" void kernel_launch(void* const* d_in, const int* in_sizes, int n_in,
                              void* d_out, int out_size, void* d_ws, size_t ws_size,
                              hipStream_t stream) {
  const float* X = (const float*)d_in[0];     // [16384, 256]
  const float* A = (const float*)d_in[1];     // [16384, 16384]
  const float* w = (const float*)d_in[2];     // [16384]
  const float* W = (const float*)d_in[3];     // [256, 256]
  float* out = (float*)d_out;

  __bf16* XT = (__bf16*)d_ws;                                          // 8 MiB
  __bf16* WT = (__bf16*)((char*)d_ws + (size_t)D_FEAT * N_NODES * 2);  // 128 KiB

  prep_xt<<<(N_NODES * D_FEAT) / 256, 256, 0, stream>>>(X, XT);
  prep_wt<<<(D_FEAT * D_FEAT) / 256, 256, 0, stream>>>(W, WT);
  gcn_fused<<<N_NODES / 16, 512, 0, stream>>>(A, X, w, XT, WT, out);
}
